// QwenAttention_heavy_hitter_54614804136283
// MI455X (gfx1250) — matrix operand order repaired and checked by reading, not compiled, not run on hardware
//
#include <hip/hip_runtime.h>
#include <hip/hip_bf16.h>
#include <math.h>

typedef __attribute__((ext_vector_type(16))) __bf16 v16bf;
typedef __attribute__((ext_vector_type(8)))  __bf16 v8bf;
typedef __attribute__((ext_vector_type(8)))  float  v8f;
typedef __attribute__((ext_vector_type(4)))  unsigned int u32x4;
typedef __attribute__((ext_vector_type(8)))  int i32x8;
typedef __attribute__((ext_vector_type(4)))  int i32x4;

#define S_LEN   2048
#define D_MODEL 2048
#define NH      16
#define NKVH    4
#define HDIM    128
#define QN      (NH * HDIM)    // 2048
#define KVN     (NKVH * HDIM)  // 512
#define HEAVY   204
#define RECENT  204
#define SEL_N   (S_LEN - RECENT)  // 1844
#define SCALE_F 0.08838834764831845f  // 128^-0.5

#if __has_builtin(__builtin_amdgcn_tensor_load_to_lds) && \
    __has_builtin(__builtin_amdgcn_s_wait_tensorcnt)
#define HAS_TDM 1
#else
#define HAS_TDM 0
#endif

static __device__ __forceinline__ v8f wmma_bf16(v16bf a, v16bf b, v8f c) {
  return __builtin_amdgcn_wmma_f32_16x16x32_bf16(false, a, false, b, (short)0, c,
                                                 false, false);
}

// A-fragment (16x32 bf16): lane holds row (lane&15); elements 0-15 = K kb..kb+15,
// where kb = (lane>>4)*16. `p` = &row[kk + kb].
static __device__ __forceinline__ v16bf load_frag_a(const __bf16* p) {
  union { v16bf v; v8bf h[2]; } u;
  u.h[0] = *(const v8bf*)p;
  u.h[1] = *(const v8bf*)(p + 8);
  return u.v;
}

#if HAS_TDM
// Issue one TDM 2D tile load (bf16 elements): tile (tile_d0 x tile_d1) from a
// row-major tensor with dim0 length tensor_d0 (elements), row pitch d0_stride
// (elements), tile_d1 rows total tensor_d1, into LDS at lds_byte_off (packed
// tile_d0*2-byte rows). D# layout per CDNA5 ISA 8.3/8.4.
static __device__ __forceinline__ void tdm_load_2d(unsigned lds_byte_off,
                                                   const __bf16* gaddr,
                                                   unsigned tensor_d0,
                                                   unsigned tensor_d1,
                                                   unsigned tile_d0,
                                                   unsigned tile_d1,
                                                   unsigned d0_stride) {
  unsigned long long ga = (unsigned long long)(size_t)gaddr;
  u32x4 g0;
  g0[0] = 1u;                                   // count=1 (valid descriptor)
  g0[1] = lds_byte_off;                         // lds_addr [63:32]
  g0[2] = (unsigned)ga;                         // global_addr [95:64]
  g0[3] = (unsigned)((ga >> 32) & 0x01FFFFFFu)  // global_addr [120:96]
          | 0x80000000u;                        // type=2 ("image") [127:126]
  i32x8 g1;
  g1[0] = (int)(1u << 16);  // workgroup_mask=0, data_size=1 (2 bytes)
  g1[1] = (int)((tensor_d0 & 0xFFFFu) << 16);               // tensor_dim0 lo16
  g1[2] = (int)(((tensor_d0 >> 16) & 0xFFFFu) |             // tensor_dim0 hi16
                ((tensor_d1 & 0xFFFFu) << 16));             // tensor_dim1 lo16
  g1[3] = (int)(((tensor_d1 >> 16) & 0xFFFFu) |             // tensor_dim1 hi16
                ((tile_d0 & 0xFFFFu) << 16));               // tile_dim0
  g1[4] = (int)(tile_d1 & 0xFFFFu);                         // tile_dim1, tile_dim2=0
  g1[5] = (int)d0_stride;                                   // tensor_dim0_stride lo32
  g1[6] = 0;                                                // stride hi16, dim1_stride lo
  g1[7] = 0;
  i32x4 z4 = {0, 0, 0, 0};
#if __clang_major__ >= 23
  i32x8 z8 = {0, 0, 0, 0, 0, 0, 0, 0};
  __builtin_amdgcn_tensor_load_to_lds(g0, g1, z4, z4, z8, 0);
#else
  __builtin_amdgcn_tensor_load_to_lds(g0, g1, z4, z4, 0);
#endif
}
#else
// Fallback: cooperative copy of a (rows x 64)-element bf16 tile into LDS.
static __device__ __forceinline__ void lds_fill(__bf16* dst, const __bf16* src,
                                                int rows, int pitch) {
  for (int idx = threadIdx.x; idx < rows * 8; idx += 256) {
    int r = idx >> 3, c = (idx & 7) * 8;
    *(v8bf*)(dst + r * 64 + c) = *(const v8bf*)(src + (size_t)r * pitch + c);
  }
}
#endif

// ---------------------------------------------------------------------------
// elementwise convert f32 -> bf16
__global__ __launch_bounds__(256) void k_convert_bf16(const float* __restrict__ in,
                                                      __bf16* __restrict__ out, int n) {
  int i = blockIdx.x * 256 + threadIdx.x;
  if (i < n) out[i] = (__bf16)in[i];
}

__global__ __launch_bounds__(256) void k_zero_f32(float* __restrict__ p, int n) {
  int i = blockIdx.x * 256 + threadIdx.x;
  if (i < n) p[i] = 0.0f;
}

// in: R x C f32  ->  out: C x R bf16   (tiled 32x32 through LDS)
__global__ __launch_bounds__(256) void k_transpose_f32_bf16(const float* __restrict__ in,
                                                            __bf16* __restrict__ out,
                                                            int R, int C) {
  __shared__ float tile[32][33];
  int tx = threadIdx.x & 31, ty = threadIdx.x >> 5;
  int cb = blockIdx.x * 32, rb = blockIdx.y * 32;
  for (int i = 0; i < 4; i++)
    tile[ty + i * 8][tx] = in[(size_t)(rb + ty + i * 8) * C + cb + tx];
  __syncthreads();
  for (int i = 0; i < 4; i++)
    out[(size_t)(cb + ty + i * 8) * R + rb + tx] = (__bf16)tile[tx][ty + i * 8];
}

// in: R x C bf16  ->  out: C x R bf16
__global__ __launch_bounds__(256) void k_transpose_bf16(const __bf16* __restrict__ in,
                                                        __bf16* __restrict__ out,
                                                        int R, int C) {
  __shared__ __bf16 tile[32][33];
  int tx = threadIdx.x & 31, ty = threadIdx.x >> 5;
  int cb = blockIdx.x * 32, rb = blockIdx.y * 32;
  for (int i = 0; i < 4; i++)
    tile[ty + i * 8][tx] = in[(size_t)(rb + ty + i * 8) * C + cb + tx];
  __syncthreads();
  for (int i = 0; i < 4; i++)
    out[(size_t)(cb + ty + i * 8) * R + rb + tx] = tile[tx][ty + i * 8];
}

// ---------------------------------------------------------------------------
// C[M,N] = A[M,K] @ Bt[N,K]^T + bias.
// Block = 8 waves = 128(M) x 64(N) C tile; wave w owns rows w*16..w*16+15.
// K-step 64, TDM (tensor_load_to_lds) double-buffered HBM->LDS, WMMA from LDS.
// grid: (M/128, N/64); block: 256.
__global__ __launch_bounds__(256) void k_gemm_wmma(const __bf16* __restrict__ A,
                                                   const __bf16* __restrict__ Bt,
                                                   const float* __restrict__ bias,
                                                   __bf16* __restrict__ outb,
                                                   float* __restrict__ outf,
                                                   int M, int N, int K) {
  __shared__ __bf16 ldsA[2][128 * 64];  // 2 x 16 KB
  __shared__ __bf16 ldsB[2][64 * 64];   // 2 x  8 KB

  const int lane = threadIdx.x & 31;
  const int wv   = threadIdx.x >> 5;
  const int hf   = lane >> 4;   // half-wave select
  const int l16  = lane & 15;
  const int rowbase = blockIdx.x * 128;
  const int nbase   = blockIdx.y * 64;

  v8f acc[4];
  for (int t = 0; t < 4; t++)
    for (int e = 0; e < 8; e++) acc[t][e] = 0.0f;

  const int nk = K >> 6;  // K / 64

#if HAS_TDM
  if (wv == 0) {
    tdm_load_2d((unsigned)(size_t)&ldsA[0][0], A + (size_t)rowbase * K,
                (unsigned)K, (unsigned)M, 64u, 128u, (unsigned)K);
    tdm_load_2d((unsigned)(size_t)&ldsB[0][0], Bt + (size_t)nbase * K,
                (unsigned)K, (unsigned)N, 64u, 64u, (unsigned)K);
  }
#endif

  for (int kt = 0; kt < nk; kt++) {
    const int cur = kt & 1;
#if HAS_TDM
    if (wv == 0) {
      if (kt + 1 < nk) {
        const int kk = (kt + 1) << 6;
        tdm_load_2d((unsigned)(size_t)&ldsA[cur ^ 1][0],
                    A + (size_t)rowbase * K + kk, (unsigned)K, (unsigned)M, 64u,
                    128u, (unsigned)K);
        tdm_load_2d((unsigned)(size_t)&ldsB[cur ^ 1][0],
                    Bt + (size_t)nbase * K + kk, (unsigned)K, (unsigned)N, 64u,
                    64u, (unsigned)K);
        __builtin_amdgcn_s_wait_tensorcnt(2);  // current buffers complete
      } else {
        __builtin_amdgcn_s_wait_tensorcnt(0);
      }
    }
    __syncthreads();  // LDS tiles `cur` visible to all waves
#else
    const int kk = kt << 6;
    __syncthreads();  // previous iteration done reading buffer 0
    lds_fill(&ldsA[0][0], A + (size_t)rowbase * K + kk, 128, K);
    lds_fill(&ldsB[0][0], Bt + (size_t)nbase * K + kk, 64, K);
    __syncthreads();
#endif

    const __bf16* at = &ldsA[HAS_TDM ? cur : 0][(wv * 16 + l16) * 64];
    const __bf16* bt = &ldsB[HAS_TDM ? cur : 0][0];
    for (int kc = 0; kc < 64; kc += 32) {
      v16bf af = load_frag_a(at + kc + hf * 16);
      for (int t = 0; t < 4; t++) {
        v16bf bfrag = *(const v16bf*)(bt + (t * 16 + l16) * 64 + kc + hf * 16);
        acc[t] = wmma_bf16(af, bfrag, acc[t]);
      }
    }
#if HAS_TDM
    __syncthreads();  // all waves done with `cur` before TDM refills it
#endif
  }

  for (int t = 0; t < 4; t++) {
    int col = nbase + t * 16 + l16;
    float bv = bias ? bias[col] : 0.0f;
    for (int r = 0; r < 8; r++) {
      int row = rowbase + wv * 16 + r + hf * 8;
      float v = acc[t][r] + bv;
      if (outb) outb[(size_t)row * N + col] = (__bf16)v;
      if (outf) outf[(size_t)row * N + col] = v;
    }
  }
}

// ---------------------------------------------------------------------------
// Pass 1: per (head, 16-query tile) wave: online softmax stats m, l.
// grid: 256 blocks x 256 threads = 2048 waves = 16 heads * 128 qtiles.
__global__ __launch_bounds__(256) void k_attn_pass1(const __bf16* __restrict__ Q,
                                                    const __bf16* __restrict__ Kmat,
                                                    float* __restrict__ mbuf,
                                                    float* __restrict__ lbuf) {
  const int lane = threadIdx.x & 31;
  const int wv   = threadIdx.x >> 5;
  const int hf   = lane >> 4;
  const int l16  = lane & 15;
  const int gw   = blockIdx.x * 8 + wv;
  const int h    = gw >> 7;
  const int qt   = gw & 127;
  const int hkv  = h >> 2;  // GROUPS = 4

  union { v16bf v; v8bf h2[2]; } qa[4];
  const __bf16* qrow = Q + (size_t)(qt * 16 + l16) * QN + h * HDIM;
  for (int d = 0; d < 4; d++) {
    const __bf16* p = qrow + d * 32 + hf * 16;
    qa[d].h2[0] = *(const v8bf*)p;
    qa[d].h2[1] = *(const v8bf*)(p + 8);
  }

  float mrow[8], lrow[8];
  for (int r = 0; r < 8; r++) { mrow[r] = -__builtin_inff(); lrow[r] = 0.0f; }

  for (int kt = 0; kt <= qt; kt++) {
    v8f c;
    for (int e = 0; e < 8; e++) c[e] = 0.0f;
    const __bf16* kbase = Kmat + (size_t)(kt * 16 + l16) * KVN + hkv * HDIM + hf * 16;
    for (int d = 0; d < 4; d++)
      c = wmma_bf16(qa[d].v, *(const v16bf*)(kbase + d * 32), c);

    int key = kt * 16 + l16;
    for (int r = 0; r < 8; r++) {
      int qq = qt * 16 + r + hf * 8;
      float sc = (key <= qq) ? c[r] * SCALE_F : -__builtin_inff();
      float tmax = sc;
      for (int mm = 1; mm < 16; mm <<= 1)
        tmax = fmaxf(tmax, __shfl_xor(tmax, mm, 32));
      float nm = fmaxf(mrow[r], tmax);
      float e  = __expf(sc - nm);
      float es = e;
      for (int mm = 1; mm < 16; mm <<= 1)
        es += __shfl_xor(es, mm, 32);
      lrow[r] = lrow[r] * __expf(mrow[r] - nm) + es;
      mrow[r] = nm;
    }
  }

  if (l16 == 0) {
    for (int r = 0; r < 8; r++) {
      int qq = qt * 16 + hf * 8 + r;
      mbuf[h * S_LEN + qq] = mrow[r];
      lbuf[h * S_LEN + qq] = lrow[r];
    }
  }
}

// ---------------------------------------------------------------------------
// Pass 2: probs, column-sums (cur_scores via atomics), O = P @ V.
__global__ __launch_bounds__(256) void k_attn_pass2(const __bf16* __restrict__ Q,
                                                    const __bf16* __restrict__ Kmat,
                                                    const __bf16* __restrict__ Vt,
                                                    const float* __restrict__ mbuf,
                                                    const float* __restrict__ lbuf,
                                                    float* __restrict__ cur_scores,
                                                    __bf16* __restrict__ attn_out) {
  __shared__ __bf16 lds_p[8][16][32];  // per-wave 16x32 P tile (bf16)

  const int lane = threadIdx.x & 31;
  const int wv   = threadIdx.x >> 5;
  const int hf   = lane >> 4;
  const int l16  = lane & 15;
  const int gw   = blockIdx.x * 8 + wv;
  const int h    = gw >> 7;
  const int qt   = gw & 127;
  const int hkv  = h >> 2;

  union { v16bf v; v8bf h2[2]; } qa[4];
  const __bf16* qrow = Q + (size_t)(qt * 16 + l16) * QN + h * HDIM;
  for (int d = 0; d < 4; d++) {
    const __bf16* p = qrow + d * 32 + hf * 16;
    qa[d].h2[0] = *(const v8bf*)p;
    qa[d].h2[1] = *(const v8bf*)(p + 8);
  }

  float mrow[8], linv[8];
  for (int r = 0; r < 8; r++) {
    int qq = qt * 16 + hf * 8 + r;
    mrow[r] = mbuf[h * S_LEN + qq];
    linv[r] = 1.0f / lbuf[h * S_LEN + qq];
  }

  v8f o[8];
  for (int dt = 0; dt < 8; dt++)
    for (int e = 0; e < 8; e++) o[dt][e] = 0.0f;

  const int qmax = qt * 16 + 15;
  const int nch  = (qt >> 1) + 1;  // 32-key chunks

  for (int ch = 0; ch < nch; ch++) {
    int kk = ch * 32;
    for (int sub = 0; sub < 2; sub++) {
      int kt16 = kk + sub * 16;
      if (kt16 <= qmax) {
        v8f c;
        for (int e = 0; e < 8; e++) c[e] = 0.0f;
        const __bf16* kbase =
            Kmat + (size_t)(kt16 + l16) * KVN + hkv * HDIM + hf * 16;
        for (int d = 0; d < 4; d++)
          c = wmma_bf16(qa[d].v, *(const v16bf*)(kbase + d * 32), c);

        int key = kt16 + l16;
        float csum = 0.0f;
        for (int r = 0; r < 8; r++) {
          int qq = qt * 16 + r + hf * 8;
          float p = (key <= qq) ? __expf(c[r] * SCALE_F - mrow[r]) * linv[r] : 0.0f;
          csum += p;
          lds_p[wv][r + hf * 8][sub * 16 + l16] = (__bf16)p;
        }
        csum += __shfl_xor(csum, 16, 32);
        if (hf == 0) atomicAdd(&cur_scores[h * S_LEN + key], csum);
      } else {
        for (int r = 0; r < 8; r++)
          lds_p[wv][r + hf * 8][sub * 16 + l16] = (__bf16)0.0f;
      }
    }

    // re-read P as an A-fragment (wave-private LDS; DS ops are in-order per wave)
    const __bf16* lp = &lds_p[wv][l16][hf * 16];
    v16bf pa = load_frag_a(lp);

    const __bf16* vrow = Vt + (size_t)(hkv * HDIM + l16) * S_LEN + kk + hf * 16;
    for (int dt = 0; dt < 8; dt++) {
      v16bf bfrag = *(const v16bf*)(vrow + (size_t)dt * 16 * S_LEN);
      o[dt] = wmma_bf16(pa, bfrag, o[dt]);
    }
  }

  for (int dt = 0; dt < 8; dt++)
    for (int r = 0; r < 8; r++) {
      int qq  = qt * 16 + r + hf * 8;
      int col = h * HDIM + dt * 16 + l16;
      attn_out[(size_t)qq * QN + col] = (__bf16)o[dt][r];
    }
}

// ---------------------------------------------------------------------------
// Per-head top-k heavy-hitter mask. grid: 16 blocks x 256 threads.
__global__ __launch_bounds__(256) void k_topk_mask(const float* __restrict__ cur,
                                                   float* __restrict__ mask) {
  __shared__ float vals[SEL_N];
  __shared__ float rv[256];
  __shared__ int   ri[256];
  const int h = blockIdx.x, tid = threadIdx.x;

  for (int i = tid; i < SEL_N; i += 256) vals[i] = cur[h * S_LEN + i];
  for (int i = tid; i <= S_LEN; i += 256)
    mask[h * (S_LEN + 1) + i] = (i >= S_LEN + 1 - RECENT) ? 1.0f : 0.0f;
  __syncthreads();

  for (int it = 0; it < HEAVY; it++) {
    float bv = -__builtin_inff();
    int bi = 0;
    for (int i = tid; i < SEL_N; i += 256) {
      float v = vals[i];
      if (v > bv) { bv = v; bi = i; }
    }
    rv[tid] = bv; ri[tid] = bi;
    __syncthreads();
    for (int s = 128; s > 0; s >>= 1) {
      if (tid < s && rv[tid + s] > rv[tid]) { rv[tid] = rv[tid + s]; ri[tid] = ri[tid + s]; }
      __syncthreads();
    }
    if (tid == 0) {
      mask[h * (S_LEN + 1) + ri[0]] = 1.0f;
      vals[ri[0]] = -__builtin_inff();
    }
    __syncthreads();
  }
}

// ---------------------------------------------------------------------------
extern "C" void kernel_launch(void* const* d_in, const int* in_sizes, int n_in,
                              void* d_out, int out_size, void* d_ws, size_t ws_size,
                              hipStream_t stream) {
  const float* hs  = (const float*)d_in[0];
  const float* q_w = (const float*)d_in[1];
  const float* q_b = (const float*)d_in[2];
  const float* k_w = (const float*)d_in[3];
  const float* k_b = (const float*)d_in[4];
  const float* v_w = (const float*)d_in[5];
  const float* v_b = (const float*)d_in[6];
  const float* o_w = (const float*)d_in[7];
  float* out_f = (float*)d_out;                         // (S, D)
  float* out_mask = out_f + (size_t)S_LEN * D_MODEL;    // (16, 2049)

  char* ws = (char*)d_ws;
  size_t off = 0;
  auto carve = [&](size_t bytes) -> char* {
    char* p = ws + off;
    off = (off + bytes + 255) & ~(size_t)255;
    return p;
  };
  __bf16* hs_b   = (__bf16*)carve((size_t)S_LEN * D_MODEL * 2);
  __bf16* qwT    = (__bf16*)carve((size_t)QN * D_MODEL * 2);
  __bf16* kwT    = (__bf16*)carve((size_t)KVN * D_MODEL * 2);
  __bf16* vwT    = (__bf16*)carve((size_t)KVN * D_MODEL * 2);
  __bf16* owT    = (__bf16*)carve((size_t)D_MODEL * QN * 2);
  __bf16* q_b16  = (__bf16*)carve((size_t)S_LEN * QN * 2);
  __bf16* k_b16  = (__bf16*)carve((size_t)S_LEN * KVN * 2);
  __bf16* v_b16  = (__bf16*)carve((size_t)S_LEN * KVN * 2);
  __bf16* vt_b16 = (__bf16*)carve((size_t)KVN * S_LEN * 2);
  __bf16* ao_b16 = (__bf16*)carve((size_t)S_LEN * QN * 2);
  float*  mbuf   = (float*)carve((size_t)NH * S_LEN * 4);
  float*  lbuf   = (float*)carve((size_t)NH * S_LEN * 4);
  float*  curS   = (float*)carve((size_t)NH * S_LEN * 4);

  // 1) hidden -> bf16
  k_convert_bf16<<<(S_LEN * D_MODEL) / 256, 256, 0, stream>>>(hs, hs_b,
                                                              S_LEN * D_MODEL);
  // 2) weights -> transposed bf16 (Wt[n,k] = W[k,n])
  k_transpose_f32_bf16<<<dim3(QN / 32, D_MODEL / 32), 256, 0, stream>>>(q_w, qwT,
                                                                        D_MODEL, QN);
  k_transpose_f32_bf16<<<dim3(KVN / 32, D_MODEL / 32), 256, 0, stream>>>(k_w, kwT,
                                                                         D_MODEL, KVN);
  k_transpose_f32_bf16<<<dim3(KVN / 32, D_MODEL / 32), 256, 0, stream>>>(v_w, vwT,
                                                                         D_MODEL, KVN);
  k_transpose_f32_bf16<<<dim3(D_MODEL / 32, QN / 32), 256, 0, stream>>>(o_w, owT,
                                                                        QN, D_MODEL);
  // 3) QKV projections (TDM-fed WMMA GEMMs, bf16 out)
  k_gemm_wmma<<<dim3(S_LEN / 128, QN / 64), 256, 0, stream>>>(
      hs_b, qwT, q_b, q_b16, (float*)nullptr, S_LEN, QN, D_MODEL);
  k_gemm_wmma<<<dim3(S_LEN / 128, KVN / 64), 256, 0, stream>>>(
      hs_b, kwT, k_b, k_b16, (float*)nullptr, S_LEN, KVN, D_MODEL);
  k_gemm_wmma<<<dim3(S_LEN / 128, KVN / 64), 256, 0, stream>>>(
      hs_b, vwT, v_b, v_b16, (float*)nullptr, S_LEN, KVN, D_MODEL);
  // 4) V -> Vt (d-major) for contiguous B-fragments in P@V
  k_transpose_bf16<<<dim3(KVN / 32, S_LEN / 32), 256, 0, stream>>>(v_b16, vt_b16,
                                                                   S_LEN, KVN);
  // 5) heavy-hitter score accumulator = 0
  k_zero_f32<<<(NH * S_LEN) / 256, 256, 0, stream>>>(curS, NH * S_LEN);
  // 6) attention
  k_attn_pass1<<<256, 256, 0, stream>>>(q_b16, k_b16, mbuf, lbuf);
  k_attn_pass2<<<256, 256, 0, stream>>>(q_b16, k_b16, vt_b16, mbuf, lbuf, curS,
                                        ao_b16);
  // 7) output projection -> f32 d_out
  k_gemm_wmma<<<dim3(S_LEN / 128, D_MODEL / 64), 256, 0, stream>>>(
      ao_b16, owT, (const float*)nullptr, (__bf16*)nullptr, out_f, S_LEN, D_MODEL,
      QN);
  // 8) heavy-hitter mask
  k_topk_mask<<<NH, 256, 0, stream>>>(curS, out_mask);
}
